// Head_39118562132225
// MI455X (gfx1250) — matrix operand order repaired and checked by reading, not compiled, not run on hardware
//
#include <hip/hip_runtime.h>
#include <math.h>

// CDNA5 / gfx1250 flash-attention head:
//   q,k,v = x @ Wq/Wk/Wv  (bf16 WMMA, f32 accumulate)
//   out   = softmax(mask(q k^T / sqrt(H))) @ v   (streaming / flash)
// Tile staging uses GLOBAL_LOAD_ASYNC_TO_LDS_B128 (ASYNCcnt); WMMA fragments
// are built from LDS; fp32->bf16 pairs use the packed v_cvt_pk_bf16_f32 path.

typedef __attribute__((ext_vector_type(16))) __bf16 v16bf;
typedef __attribute__((ext_vector_type(2)))  __bf16 v2bf;
typedef __attribute__((ext_vector_type(8)))  float  v8f;

namespace {

constexpr int kB = 16;
constexpr int kT = 2048;
constexpr int kC = 384;
constexpr int kH = 64;

union ABFrag {
  v16bf v;
  unsigned u[8];
};
static_assert(sizeof(ABFrag) == 32, "frag must be 8 VGPRs");

// dword load of a bf16 pair via memcpy (no TBAA reordering vs __bf16 stores).
__device__ __forceinline__ unsigned ldpair(const __bf16* p) {
  unsigned u;
  __builtin_memcpy(&u, p, 4);
  return u;
}

// pack two bf16 into one dword (low = first) via native 2-vector.
__device__ __forceinline__ unsigned pk2(__bf16 lo, __bf16 hi) {
  v2bf p;
  p.x = lo;
  p.y = hi;
  return __builtin_bit_cast(unsigned, p);
}

// convert two f32 to bf16 (RNE) and pack -> v_cvt_pk_bf16_f32.
__device__ __forceinline__ unsigned pkbf(float a, float b) {
  v2bf p;
  p.x = (__bf16)a;
  p.y = (__bf16)b;
  return __builtin_bit_cast(unsigned, p);
}

// Async DMA: 16 bytes per lane, global -> LDS, tracked by ASYNCcnt.
// vdst operand carries the LDS byte address (low 32 bits of the generic ptr).
__device__ __forceinline__ void async_b128(const void* gptr, void* lptr) {
  unsigned lds = (unsigned)(size_t)lptr;
  asm volatile("global_load_async_to_lds_b128 %0, %1, off"
               :
               : "v"(lds), "v"(gptr)
               : "memory");
}
__device__ __forceinline__ void wait_async0() {
  asm volatile("s_wait_asynccnt 0x0" ::: "memory");
}

}  // namespace

// ---------------------------------------------------------------------------
// Kernel 1: QKV projection.  grid = B * T/64 blocks of 128 threads (4 waves).
// x/W tiles DMA'd to LDS as fp32 (b128 async), converted to bf16 while
// building WMMA fragments.  12 v_wmma_f32_16x16x32_bf16 per K-step per wave.
// ---------------------------------------------------------------------------
__global__ __launch_bounds__(128) void qkv_proj_kernel(
    const float* __restrict__ x, const float* __restrict__ Wq,
    const float* __restrict__ Wk, const float* __restrict__ Wv,
    unsigned short* __restrict__ qg_, unsigned short* __restrict__ kg_,
    unsigned short* __restrict__ vg_) {
  __bf16* qg = (__bf16*)qg_;
  __bf16* kg = (__bf16*)kg_;
  __bf16* vg = (__bf16*)vg_;

  __shared__ float Wf[3][32 * 64];  // [c][h], rows 256 B (b128 aligned)
  __shared__ float Xf[4][16 * 32];  // per-wave [m][c], rows 128 B

  const int tid = threadIdx.x;
  const int wave = tid >> 5;
  const int lane = tid & 31;
  const int m = lane & 15;  // A row / B column within 16
  const int g = lane >> 4;  // 16-lane half

  const int b = blockIdx.x >> 5;          // T/64 = 32 tiles per batch
  const int tb = (blockIdx.x & 31) * 64;  // first T row of block
  const int trow0 = tb + wave * 16;       // first T row of wave

  v8f acc[3][4] = {};  // [Wq/Wk/Wv][N-tile of 16]

  const float* Ws[3] = {Wq, Wk, Wv};

  for (int c0 = 0; c0 < kC; c0 += 32) {
    // Async stage W chunk: 3 x 32 x 64 f32 = 1536 b128 chunks / 128 threads.
    for (int i = 0; i < 12; ++i) {
      int c = tid + 128 * i;  // 0..1535
      int w = c >> 9;
      int r = c & 511;
      int cl = r >> 4;        // 0..31
      int h0 = (r & 15) * 4;  // 0..60
      async_b128(&Ws[w][(size_t)(c0 + cl) * kH + h0], &Wf[w][cl * 64 + h0]);
    }
    // Async stage this wave's x tile: 16 x 32 f32 = 128 chunks / 32 lanes.
    for (int i = 0; i < 4; ++i) {
      int c = lane + 32 * i;  // 0..127
      int m2 = c >> 3;
      int c4 = (c & 7) * 4;
      async_b128(&x[((size_t)b * kT + trow0 + m2) * kC + c0 + c4],
                 &Xf[wave][m2 * 32 + c4]);
    }
    wait_async0();
    __syncthreads();

    // A fragment (16x32 bf16): lane holds row m, K order per ISA table.
    ABFrag a;
    for (int j = 0; j < 8; ++j) {
      int k = ((j >> 2) << 4) + g * 8 + ((j & 3) << 1);
      a.u[j] = pkbf(Xf[wave][m * 32 + k], Xf[wave][m * 32 + k + 1]);
    }

    for (int w = 0; w < 3; ++w) {
      for (int nt = 0; nt < 4; ++nt) {
        // B fragment: lane holds column n = 16*nt + m; K pairs over c.
        ABFrag bf;
        int n = nt * 16 + m;
        for (int j = 0; j < 8; ++j) {
          int cc = ((j >> 2) << 4) + g * 8 + ((j & 3) << 1);
          bf.u[j] = pkbf(Wf[w][cc * 64 + n], Wf[w][(cc + 1) * 64 + n]);
        }
        acc[w][nt] = __builtin_amdgcn_wmma_f32_16x16x32_bf16(
            false, a.v, false, bf.v, (short)0, acc[w][nt], false, false);
      }
    }
    __syncthreads();
  }

  // Store q/k/v tiles as bf16 [B,T,H].  C-layout: VGPR r holds row r+8g.
  __bf16* outs[3] = {qg, kg, vg};
  for (int w = 0; w < 3; ++w)
    for (int nt = 0; nt < 4; ++nt)
      for (int r = 0; r < 8; ++r) {
        int row = r + 8 * g;
        int h = nt * 16 + m;
        outs[w][((size_t)b * kT + trow0 + row) * kH + h] =
            (__bf16)acc[w][nt][r];
      }
}

// ---------------------------------------------------------------------------
// Kernel 2: streaming causal attention.  grid = B * T/64 blocks of 128
// threads; each wave owns 16 query rows, block shares K/V tiles of 32 keys
// staged with async b128 DMA.
// ---------------------------------------------------------------------------
__global__ __launch_bounds__(128) void flash_attn_kernel(
    const unsigned short* __restrict__ qg_,
    const unsigned short* __restrict__ kg_,
    const unsigned short* __restrict__ vg_, float* __restrict__ out) {
  const __bf16* qg = (const __bf16*)qg_;
  const __bf16* kg = (const __bf16*)kg_;
  const __bf16* vg = (const __bf16*)vg_;

  __shared__ __bf16 Kl[32 * 64];     // [s][h], rows 128 B (b128 aligned)
  __shared__ __bf16 Vl[32 * 64];     // [s][h], rows 128 B
  __shared__ __bf16 Pl[4][16 * 34];  // per-wave P round-trip, [m][k], padded

  const int tid = threadIdx.x;
  const int wave = tid >> 5;
  const int lane = tid & 31;
  const int m = lane & 15;
  const int g = lane >> 4;

  const int b = blockIdx.x >> 5;
  const int qb0 = (blockIdx.x & 31) * 64;  // first query row of block
  const int qt0 = qb0 + wave * 16;         // first query row of wave

  // Load Q fragments (16 rows x 64 h = two K=32 A-fragments), kept resident.
  ABFrag qa[2];
  for (int p = 0; p < 2; ++p)
    for (int j = 0; j < 8; ++j) {
      int h = p * 32 + ((j >> 2) << 4) + g * 8 + ((j & 3) << 1);
      qa[p].u[j] = ldpair(&qg[((size_t)b * kT + qt0 + m) * kH + h]);
    }

  float mi[8], li[8];
  for (int r = 0; r < 8; ++r) {
    mi[r] = -INFINITY;
    li[r] = 0.0f;
  }
  v8f O[4] = {};

  const int sEnd = qb0 + 64;  // causal: keys beyond the block are all masked
  for (int s0 = 0; s0 < sEnd; s0 += 32) {
    // Async stage K and V tiles (each 32x64 bf16 = 256 b128 chunks).
    for (int i = 0; i < 2; ++i) {
      int c = tid + 128 * i;  // 0..255
      int s = c >> 3;
      int h0 = (c & 7) * 8;
      size_t row = (size_t)b * kT + s0 + s;
      async_b128(&kg[row * kH + h0], &Kl[s * 64 + h0]);
      async_b128(&vg[row * kH + h0], &Vl[s * 64 + h0]);
    }
    wait_async0();
    __syncthreads();

    // S = Q @ K^T for two 16x16 key tiles (K-dim = H = 64 -> 2 WMMAs each).
    v8f S[2];
    for (int t = 0; t < 2; ++t) {
      v8f sacc = {};
      for (int p = 0; p < 2; ++p) {
        ABFrag kf;
        for (int j = 0; j < 8; ++j) {
          int h = p * 32 + ((j >> 2) << 4) + g * 8 + ((j & 3) << 1);
          kf.u[j] = ldpair(&Kl[(t * 16 + m) * 64 + h]);
        }
        sacc = __builtin_amdgcn_wmma_f32_16x16x32_bf16(
            false, qa[p].v, false, kf.v, (short)0, sacc, false, false);
      }
      S[t] = sacc;
    }

    // Online softmax.  Row r+8g lives in the lane's own 16-lane half; xor
    // shuffles 1/2/4/8 reduce across the 16 lanes holding that row.
    float p0[8], p1[8];
    for (int r = 0; r < 8; ++r) {
      int qrow = qt0 + r + 8 * g;
      float v0 = S[0][r] * 0.125f;  // 1/sqrt(64)
      float v1 = S[1][r] * 0.125f;
      if (s0 + m > qrow) v0 = -INFINITY;       // causal mask, tile 0
      if (s0 + 16 + m > qrow) v1 = -INFINITY;  // causal mask, tile 1
      float mx = fmaxf(v0, v1);
      mx = fmaxf(mx, __shfl_xor(mx, 1, 32));
      mx = fmaxf(mx, __shfl_xor(mx, 2, 32));
      mx = fmaxf(mx, __shfl_xor(mx, 4, 32));
      mx = fmaxf(mx, __shfl_xor(mx, 8, 32));
      float mnew = fmaxf(mi[r], mx);
      float corr = __expf(mi[r] - mnew);  // first iter: exp(-inf)=0
      float e0 = __expf(v0 - mnew);
      float e1 = __expf(v1 - mnew);
      float rs = e0 + e1;
      rs += __shfl_xor(rs, 1, 32);
      rs += __shfl_xor(rs, 2, 32);
      rs += __shfl_xor(rs, 4, 32);
      rs += __shfl_xor(rs, 8, 32);
      li[r] = li[r] * corr + rs;
      mi[r] = mnew;
      p0[r] = e0;
      p1[r] = e1;
      for (int nt = 0; nt < 4; ++nt) O[nt][r] *= corr;
    }

    // Round-trip P through per-wave LDS: C-layout f32 -> A-layout bf16.
    for (int r = 0; r < 8; ++r) {
      int row = r + 8 * g;
      Pl[wave][row * 34 + m] = (__bf16)p0[r];
      Pl[wave][row * 34 + 16 + m] = (__bf16)p1[r];
    }
    ABFrag pf;  // same-wave LDS ops are in-order; waits inserted by compiler
    for (int j = 0; j < 8; ++j) {
      int k = ((j >> 2) << 4) + g * 8 + ((j & 3) << 1);
      pf.u[j] = ldpair(&Pl[wave][m * 34 + k]);
    }

    // O += P @ V.  V B-fragment: K pairs stride over s -> two b16 loads+pack.
    for (int nt = 0; nt < 4; ++nt) {
      ABFrag vf;
      int h = nt * 16 + m;
      for (int j = 0; j < 8; ++j) {
        int s = ((j >> 2) << 4) + g * 8 + ((j & 3) << 1);
        vf.u[j] = pk2(Vl[s * 64 + h], Vl[(s + 1) * 64 + h]);
      }
      O[nt] = __builtin_amdgcn_wmma_f32_16x16x32_bf16(
          false, pf.v, false, vf.v, (short)0, O[nt], false, false);
    }
    __syncthreads();
  }

  // out = O / l, fp32 [B,T,H].
  for (int nt = 0; nt < 4; ++nt)
    for (int r = 0; r < 8; ++r) {
      int row = r + 8 * g;
      out[((size_t)b * kT + qt0 + row) * kH + nt * 16 + m] =
          O[nt][r] / li[r];
    }
}

// ---------------------------------------------------------------------------
extern "C" void kernel_launch(void* const* d_in, const int* in_sizes, int n_in,
                              void* d_out, int out_size, void* d_ws,
                              size_t ws_size, hipStream_t stream) {
  (void)in_sizes;
  (void)n_in;
  (void)out_size;
  (void)ws_size;

  const float* x = (const float*)d_in[0];
  const float* Wq = (const float*)d_in[1];
  const float* Wk = (const float*)d_in[2];
  const float* Wv = (const float*)d_in[3];
  float* out = (float*)d_out;

  // Workspace: q/k/v as bf16 [B,T,H] -> 3 * 4 MB = 12 MB.
  const size_t nqkv = (size_t)kB * kT * kH;
  unsigned short* qws = (unsigned short*)d_ws;
  unsigned short* kws = qws + nqkv;
  unsigned short* vws = kws + nqkv;

  const int blocks = kB * (kT / 64);  // 512
  qkv_proj_kernel<<<blocks, 128, 0, stream>>>(x, Wq, Wk, Wv, qws, kws, vws);
  flash_attn_kernel<<<blocks, 128, 0, stream>>>(qws, kws, vws, out);
}
